// Obs_IN_10634339025628
// MI455X (gfx1250) — matrix operand order repaired and checked by reading, not compiled, not run on hardware
//
#include <hip/hip_runtime.h>

// ---------------------------------------------------------------------------
// CDNA5 (gfx1250) wave32 WMMA implementation of the obs/comm/MTRNN pipeline.
// All GEMM-shaped math uses v_wmma_f32_16x16x32_bf16 (f32 accumulate).
// The RNN kernel stages its per-step activation tile with CDNA5 async
// global->LDS DMA (global_load_async_to_lds_b128 + s_wait_asynccnt).
// ---------------------------------------------------------------------------

typedef __attribute__((ext_vector_type(16))) __bf16 v16bf;
typedef __attribute__((ext_vector_type(8)))  __bf16 v8bf;
typedef __attribute__((ext_vector_type(8)))  float  v8f;

#define NSAMP 4096      // E*S
#define H     256
#define LSTEP 32
#define KOBJ  32
#define VOBS  128
#define VCOMM 32
#define OBS_K (KOBJ * H)   // 8192

// LDS row strides (in bf16 halves) padded to avoid 64-bank conflicts
#define ESTR  264
#define CESTR 264
#define BSTR  264          // 264*2 = 528 B rows, 16B aligned
#define HSTR  260          // f32 stride

__device__ __forceinline__ v8f wmma_bf16(v16bf a, v16bf b, v8f c) {
  return __builtin_amdgcn_wmma_f32_16x16x32_bf16(
      /*neg_a=*/false, a, /*neg_b=*/false, b,
      /*c_mod=*/(short)0, c, /*reuse_a=*/false, /*reuse_b=*/false);
}

// A fragment (16x32 bf16): lane l<16 holds row m=l with K {0..15},
// lane l>=16 holds row m=l-16 with K {16..31}. rowptr -> 32 contiguous K.
__device__ __forceinline__ v16bf load_a_frag(const __bf16* rowptr, int lane) {
  const int base = (lane < 16) ? 0 : 16;
  v8bf lo = *(const v8bf*)(rowptr + base);
  v8bf hi = *(const v8bf*)(rowptr + base + 8);
  v16bf a;
#pragma unroll
  for (int i = 0; i < 8; ++i) { a[i] = lo[i]; a[8 + i] = hi[i]; }
  return a;
}

// B fragment (32x16 bf16) from an N-major (transposed) weight matrix:
// lane n holds column n&15; lanes 0-15 K=0..15, lanes 16-31 K=16..31.
__device__ __forceinline__ v16bf load_b_frag(const __bf16* colptr, int lane) {
  return *(const v16bf*)(colptr + ((lane < 16) ? 0 : 16));
}

__device__ __forceinline__ float preluf(float x, float a) {
  return x >= 0.f ? x : a * x;
}
__device__ __forceinline__ float sigm(float x) {
  return 1.f / (1.f + __expf(-x));
}

// ---------------------------------------------------------------------------
// Weight conversion kernels (f32 -> bf16, transposed to N-major)
// ---------------------------------------------------------------------------
__global__ __launch_bounds__(256) void k_transpose(const float* __restrict__ src,
                                                   __bf16* __restrict__ dst,
                                                   int R, int C) {
  int i = blockIdx.x * 256 + threadIdx.x;
  if (i >= R * C) return;
  int r = i / C, c = i % C;
  dst[(size_t)c * R + r] = (__bf16)src[i];   // dst[N][K]
}

// cw[co][ci][dk]  ->  dst[co][dk*256+ci]   (N-major over K = dk*256+ci)
__global__ __launch_bounds__(256) void k_convw(const float* __restrict__ src,
                                               __bf16* __restrict__ dst,
                                               int ktap) {
  int total = 64 * 256 * ktap;
  int i = blockIdx.x * 256 + threadIdx.x;
  if (i >= total) return;
  int co  = i / (256 * ktap);
  int rem = i - co * 256 * ktap;
  int ci  = rem / ktap;
  int dk  = rem - ci * ktap;
  dst[(size_t)co * 256 * ktap + dk * 256 + ci] = (__bf16)src[i];
}

// ---------------------------------------------------------------------------
// Obs path: gather+PReLU of 32 embedding rows per sample, then
// [4096 x 8192] x [8192 x 256] WMMA GEMM + bias + PReLU -> out[:, 0:256]
// ---------------------------------------------------------------------------
__global__ __launch_bounds__(256) void k_obs(
    const float* __restrict__ obs, const float* __restrict__ obs_emb,
    const float* __restrict__ p_aemb, const __bf16* __restrict__ WobsT,
    const float* __restrict__ obs_b, const float* __restrict__ p_a,
    float* __restrict__ out) {
  __shared__ alignas(32) __bf16 emb[VOBS * ESTR];
  __shared__ int idxs[16][KOBJ];
  const int t = threadIdx.x;
  const float aemb = *p_aemb;

  for (int i = t; i < VOBS * H; i += 256) {
    int r = i >> 8, c = i & 255;
    emb[r * ESTR + c] = (__bf16)preluf(obs_emb[i], aemb);
  }
  const int n0 = blockIdx.x * 16;
  if (t < 16) {
    const float* orow = obs + (size_t)(n0 + t) * VOBS;
    int cnt = 0;
    for (int v = 0; v < VOBS; ++v)
      if (orow[v] > 0.5f && cnt < KOBJ) idxs[t][cnt++] = v;
  }
  __syncthreads();

  const int wave = t >> 5, lane = t & 31, m = lane & 15;
  const float a_o = *p_a;
#pragma unroll
  for (int nti = 0; nti < 2; ++nti) {
    const int nt  = wave * 2 + nti;
    const int col = nt * 16 + m;
    const __bf16* bcol = WobsT + (size_t)col * OBS_K;
    v8f acc = {};
    for (int kc = 0; kc < OBS_K / 32; ++kc) {   // 256 K-chunks of 32
      if ((kc & 15) == 0)
        __builtin_prefetch(bcol + (kc + 16) * 32, 0, 1);
      const int slot = kc >> 3;
      const int h0   = (kc & 7) << 5;
      const __bf16* ap = emb + idxs[m][slot] * ESTR + h0;
      v16bf a = load_a_frag(ap, lane);
      v16bf b = load_b_frag(bcol + kc * 32, lane);
      acc = wmma_bf16(a, b, acc);
    }
    const float bias = obs_b[col];
#pragma unroll
    for (int v = 0; v < 8; ++v) {
      const int row = v + ((lane < 16) ? 0 : 8);
      out[(size_t)(n0 + row) * 512 + col] = preluf(acc[v] + bias, a_o);
    }
  }
}

// ---------------------------------------------------------------------------
// Comm conv path: argmax tokens -> embed+PReLU into LDS, then the four
// 1-D convs as implicit WMMA GEMMs; writes y[n][l][ch] bf16 and per-block
// per-channel (sum, sumsq) partials for BN.
// ---------------------------------------------------------------------------
__global__ __launch_bounds__(256) void k_conv(
    const float* __restrict__ comm, const float* __restrict__ comm_emb,
    const float* __restrict__ p_aemb,
    const __bf16* __restrict__ Wc1, const __bf16* __restrict__ Wc3,
    const __bf16* __restrict__ Wc5, const __bf16* __restrict__ Wc7,
    const float* __restrict__ cb1, const float* __restrict__ cb3,
    const float* __restrict__ cb5, const float* __restrict__ cb7,
    __bf16* __restrict__ yg, float* __restrict__ psumS,
    float* __restrict__ psumQ) {
  __shared__ alignas(32) __bf16 ce[33 * CESTR];   // row 32 = zeros (padding row)
  __shared__ int tok[LSTEP];
  const int t = threadIdx.x;
  const int n = blockIdx.x;

  if (t < LSTEP) {                                // argmax over V_COMM
    const float* crow = comm + ((size_t)n * LSTEP + t) * VCOMM;
    float best = crow[0]; int bi = 0;
    for (int v = 1; v < VCOMM; ++v) {
      float x = crow[v];
      if (x > best) { best = x; bi = v; }
    }
    tok[t] = bi;
  }
  for (int i = t; i < CESTR; i += 256) ce[32 * CESTR + i] = (__bf16)0.f;
  __syncthreads();
  const float aemb = *p_aemb;
  for (int i = t; i < LSTEP * H; i += 256) {
    int l = i >> 8, h = i & 255;
    ce[l * CESTR + h] = (__bf16)preluf(comm_emb[tok[l] * H + h], aemb);
  }
  __syncthreads();

  const int wave = t >> 5, lane = t & 31, m = lane & 15;
  const int ktab[4] = {1, 3, 5, 7};
  for (int half = 0; half < 2; ++half) {
    const int ct = 2 * wave + half;               // 0..15 channel tile
    const int br = ct >> 2, cosub = ct & 3;
    const __bf16* Wb = (br == 0) ? Wc1 : (br == 1) ? Wc3 : (br == 2) ? Wc5 : Wc7;
    const float*  cb = (br == 0) ? cb1 : (br == 1) ? cb3 : (br == 2) ? cb5 : cb7;
    const int ktap = ktab[br];
    const int pad  = ktap >> 1;
    const int co   = cosub * 16 + m;
    const int ch   = br * 64 + co;
    const __bf16* bcol = Wb + (size_t)co * 256 * ktap;
    float s = 0.f, q = 0.f;
    for (int mt = 0; mt < 2; ++mt) {
      v8f acc = {};
      for (int dk = 0; dk < ktap; ++dk) {
        const int inrow = mt * 16 + m + dk - pad;
        const int rsel  = (inrow >= 0 && inrow < 32) ? inrow : 32;  // zero row
        const __bf16* arow = ce + rsel * CESTR;
#pragma unroll
        for (int hc = 0; hc < 8; ++hc) {
          v16bf a = load_a_frag(arow + hc * 32, lane);
          v16bf b = load_b_frag(bcol + dk * 256 + hc * 32, lane);
          acc = wmma_bf16(a, b, acc);
        }
      }
      const float bias = cb[co];
#pragma unroll
      for (int v = 0; v < 8; ++v) {
        const int lrow = mt * 16 + v + ((lane < 16) ? 0 : 8);
        const float y = acc[v] + bias;
        yg[((size_t)n * LSTEP + lrow) * H + ch] = (__bf16)y;
        s += y; q += y * y;
      }
    }
    // combine the two lane-halves (same channel) via ds-permute shuffle
    s += __shfl_xor(s, 16, 32);
    q += __shfl_xor(q, 16, 32);
    if (lane < 16) {
      psumS[(size_t)n * H + ch] = s;
      psumQ[(size_t)n * H + ch] = q;
    }
  }
}

// Deterministic per-channel reduction of the 4096 block partials.
__global__ __launch_bounds__(256) void k_bnred(const float* __restrict__ psumS,
                                               const float* __restrict__ psumQ,
                                               float* __restrict__ bn_sum,
                                               float* __restrict__ bn_sq) {
  __shared__ float sS[256], sQ[256];
  const int ch = blockIdx.x, t = threadIdx.x;
  float s = 0.f, q = 0.f;
  for (int nrow = t; nrow < NSAMP; nrow += 256) {
    s += psumS[(size_t)nrow * H + ch];
    q += psumQ[(size_t)nrow * H + ch];
  }
  sS[t] = s; sQ[t] = q;
  __syncthreads();
  for (int st = 128; st > 0; st >>= 1) {
    if (t < st) { sS[t] += sS[t + st]; sQ[t] += sQ[t + st]; }
    __syncthreads();
  }
  if (t == 0) { bn_sum[ch] = sS[0]; bn_sq[ch] = sQ[0]; }
}

// BatchNorm (training-mode batch stats) + PReLU, in place on yg (bf16).
__global__ __launch_bounds__(256) void k_bn(__bf16* __restrict__ yg,
                                            const float* __restrict__ bn_sum,
                                            const float* __restrict__ bn_sq,
                                            const float* __restrict__ bn_g,
                                            const float* __restrict__ bn_b,
                                            const float* __restrict__ p_a) {
  const float invNL = 1.0f / ((float)NSAMP * (float)LSTEP);
  const float a = *p_a;
  size_t i0 = ((size_t)blockIdx.x * 256 + threadIdx.x) * 8;
  if (i0 >= (size_t)NSAMP * LSTEP * H) return;
#pragma unroll
  for (int j = 0; j < 8; ++j) {
    const size_t i = i0 + j;
    const int ch = (int)(i & 255);
    const float mean = bn_sum[ch] * invNL;
    const float var  = bn_sq[ch] * invNL - mean * mean;
    const float sc   = bn_g[ch] * rsqrtf(var + 1e-5f);
    float y = (float)yg[i];
    y = sc * (y - mean) + bn_b[ch];
    yg[i] = (__bf16)preluf(y, a);
  }
}

// ---------------------------------------------------------------------------
// MTRNN: each block owns 16 samples, keeps h in LDS, runs all 32 steps with
// 6 WMMA GEMMs/step (xr,xz,xn and recurrent r,z,n), then the fused final
// PReLU-linear-PReLU -> out[:, 256:512].
// The y[:,l,:] tile is staged into LDS with CDNA5 async global->LDS DMA.
// ---------------------------------------------------------------------------
__global__ __launch_bounds__(256) void k_rnn(
    const __bf16* __restrict__ yg, const __bf16* __restrict__ WxT,
    const __bf16* __restrict__ WhT, const __bf16* __restrict__ linT,
    const float* __restrict__ bx, const float* __restrict__ bh,
    const float* __restrict__ lin_b, const float* __restrict__ p_a1,
    const float* __restrict__ p_a2, float* __restrict__ out) {
  __shared__ float h[16 * HSTR];
  __shared__ alignas(32) __bf16 hb[16 * BSTR];
  __shared__ alignas(32) __bf16 yl[16 * BSTR];
  __shared__ float gr[16 * 256], gz[16 * 256], gxn[16 * 256], ghn[16 * 256];
  const int t = threadIdx.x, wave = t >> 5, lane = t & 31, m = lane & 15;
  const int n0 = blockIdx.x * 16;

  for (int i = t; i < 16 * HSTR; i += 256) h[i] = 0.f;
  __syncthreads();

  for (int l = 0; l < LSTEP; ++l) {
    // Stage y[:,l,:] (16 rows x 256 ch bf16 = 16 x 512 B) into LDS with
    // async DMA: 512 16-byte chunks, 2 per thread.  Tracked by ASYNCcnt.
    {
#pragma unroll
      for (int rep = 0; rep < 2; ++rep) {
        const unsigned i = (unsigned)t + (unsigned)rep * 256u;   // 0..511
        const unsigned mm = i >> 5, c16 = i & 31;
        const unsigned goff =
            ((((unsigned)(n0 + (int)mm)) * LSTEP + (unsigned)l) * H + c16 * 8) * 2u;
        const unsigned ldsa =
            (unsigned)(size_t)(const void*)(yl + mm * BSTR + c16 * 8);
        asm volatile("global_load_async_to_lds_b128 %0, %1, %2 offset:0"
                     :
                     : "v"(ldsa), "v"(goff), "s"(yg)
                     : "memory");
      }
    }
    for (int i = t; i < 16 * H; i += 256) {
      const int mm = i >> 8, c = i & 255;
      hb[mm * BSTR + c] = (__bf16)h[mm * HSTR + c];
    }
    asm volatile("s_wait_asynccnt 0x0" ::: "memory");
    __syncthreads();
    // 48 jobs = 3 gates x 16 N-tiles; each job = x-proj and h-proj tile
    for (int j = 0; j < 6; ++j) {
      const int job = wave + j * 8;
      const int g = job / 16, nt = job % 16;
      const int col = nt * 16 + m;
      const __bf16* bxc = WxT + ((size_t)g * H + col) * H;
      const __bf16* bhc = WhT + ((size_t)g * H + col) * H;
      v8f ax = {}, ah = {};
#pragma unroll
      for (int kc = 0; kc < 8; ++kc) {
        v16bf aY = load_a_frag(yl + m * BSTR + kc * 32, lane);
        v16bf bX = load_b_frag(bxc + kc * 32, lane);
        ax = wmma_bf16(aY, bX, ax);
        v16bf aH = load_a_frag(hb + m * BSTR + kc * 32, lane);
        v16bf bH = load_b_frag(bhc + kc * 32, lane);
        ah = wmma_bf16(aH, bH, ah);
      }
      const float bbx = bx[g * H + col], bbh = bh[g * H + col];
#pragma unroll
      for (int v = 0; v < 8; ++v) {
        const int row = v + ((lane < 16) ? 0 : 8);
        const float xv = ax[v] + bbx, hv = ah[v] + bbh;
        if (g == 0)      gr[row * 256 + col] = xv + hv;
        else if (g == 1) gz[row * 256 + col] = xv + hv;
        else { gxn[row * 256 + col] = xv; ghn[row * 256 + col] = hv; }
      }
    }
    __syncthreads();
    for (int i = t; i < 16 * H; i += 256) {
      const int mm = i >> 8, c = i & 255;
      const float r  = sigm(gr[i]);
      const float z  = sigm(gz[i]);
      const float nn = tanhf(gxn[i] + r * ghn[i]);
      h[mm * HSTR + c] = nn * (1.f - z) + h[mm * HSTR + c] * z;
    }
    __syncthreads();
  }

  // final: c = prelu(prelu(h,a1) @ lin_W + lin_b, a2)
  const float a1 = *p_a1, a2 = *p_a2;
  for (int i = t; i < 16 * H; i += 256) {
    const int mm = i >> 8, c = i & 255;
    hb[mm * BSTR + c] = (__bf16)preluf(h[mm * HSTR + c], a1);
  }
  __syncthreads();
#pragma unroll
  for (int nti = 0; nti < 2; ++nti) {
    const int nt = wave * 2 + nti;
    const int col = nt * 16 + m;
    const __bf16* bc = linT + (size_t)col * H;
    v8f acc = {};
#pragma unroll
    for (int kc = 0; kc < 8; ++kc) {
      v16bf a = load_a_frag(hb + m * BSTR + kc * 32, lane);
      v16bf b = load_b_frag(bc + kc * 32, lane);
      acc = wmma_bf16(a, b, acc);
    }
    const float bias = lin_b[col];
#pragma unroll
    for (int v = 0; v < 8; ++v) {
      const int row = v + ((lane < 16) ? 0 : 8);
      out[(size_t)(n0 + row) * 512 + 256 + col] = preluf(acc[v] + bias, a2);
    }
  }
}

// ---------------------------------------------------------------------------
extern "C" void kernel_launch(void* const* d_in, const int* in_sizes, int n_in,
                              void* d_out, int out_size, void* d_ws,
                              size_t ws_size, hipStream_t stream) {
  (void)in_sizes; (void)n_in; (void)out_size; (void)ws_size;
  const float* obs      = (const float*)d_in[0];
  const float* comm     = (const float*)d_in[1];
  const float* obs_emb  = (const float*)d_in[2];
  const float* obs_aemb = (const float*)d_in[3];
  const float* obs_W    = (const float*)d_in[4];
  const float* obs_b    = (const float*)d_in[5];
  const float* obs_a    = (const float*)d_in[6];
  const float* comm_emb = (const float*)d_in[7];
  const float* comm_aemb= (const float*)d_in[8];
  const float* cw1 = (const float*)d_in[9];  const float* cb1 = (const float*)d_in[10];
  const float* cw3 = (const float*)d_in[11]; const float* cb3 = (const float*)d_in[12];
  const float* cw5 = (const float*)d_in[13]; const float* cb5 = (const float*)d_in[14];
  const float* cw7 = (const float*)d_in[15]; const float* cb7 = (const float*)d_in[16];
  const float* bn_g = (const float*)d_in[17]; const float* bn_b = (const float*)d_in[18];
  const float* cnn_a = (const float*)d_in[19];
  const float* Wx = (const float*)d_in[20]; const float* bx = (const float*)d_in[21];
  const float* Wh = (const float*)d_in[22]; const float* bh = (const float*)d_in[23];
  const float* lin_a1 = (const float*)d_in[24];
  const float* lin_W  = (const float*)d_in[25];
  const float* lin_b  = (const float*)d_in[26];
  const float* lin_a2 = (const float*)d_in[27];
  float* out = (float*)d_out;

  char* ws = (char*)d_ws;
  size_t off = 0;
  auto alloc = [&](size_t bytes) -> void* {
    void* p = ws + off;
    off = (off + bytes + 255) & ~(size_t)255;
    return p;
  };
  __bf16* WobsT = (__bf16*)alloc((size_t)OBS_K * H * 2);     // 4 MiB
  __bf16* WxT   = (__bf16*)alloc((size_t)3 * H * H * 2);
  __bf16* WhT   = (__bf16*)alloc((size_t)3 * H * H * 2);
  __bf16* linT  = (__bf16*)alloc((size_t)H * H * 2);
  __bf16* Wc1   = (__bf16*)alloc((size_t)64 * 256 * 1 * 2);
  __bf16* Wc3   = (__bf16*)alloc((size_t)64 * 256 * 3 * 2);
  __bf16* Wc5   = (__bf16*)alloc((size_t)64 * 256 * 5 * 2);
  __bf16* Wc7   = (__bf16*)alloc((size_t)64 * 256 * 7 * 2);
  __bf16* yg    = (__bf16*)alloc((size_t)NSAMP * LSTEP * H * 2);  // 64 MiB
  float*  psumS = (float*)alloc((size_t)NSAMP * H * 4);           // 4 MiB
  float*  psumQ = (float*)alloc((size_t)NSAMP * H * 4);           // 4 MiB
  float*  bnS   = (float*)alloc(H * 4);
  float*  bnQ   = (float*)alloc(H * 4);

  // Weight conversion (bf16, N-major)
  k_transpose<<<(OBS_K * H + 255) / 256, 256, 0, stream>>>(obs_W, WobsT, OBS_K, H);
  for (int g = 0; g < 3; ++g) {
    k_transpose<<<(H * H + 255) / 256, 256, 0, stream>>>(Wx + g * H * H, WxT + g * H * H, H, H);
    k_transpose<<<(H * H + 255) / 256, 256, 0, stream>>>(Wh + g * H * H, WhT + g * H * H, H, H);
  }
  k_transpose<<<(H * H + 255) / 256, 256, 0, stream>>>(lin_W, linT, H, H);
  k_convw<<<(64 * 256 * 1 + 255) / 256, 256, 0, stream>>>(cw1, Wc1, 1);
  k_convw<<<(64 * 256 * 3 + 255) / 256, 256, 0, stream>>>(cw3, Wc3, 3);
  k_convw<<<(64 * 256 * 5 + 255) / 256, 256, 0, stream>>>(cw5, Wc5, 5);
  k_convw<<<(64 * 256 * 7 + 255) / 256, 256, 0, stream>>>(cw7, Wc7, 7);

  // Obs path -> out[:, 0:256]
  k_obs<<<NSAMP / 16, 256, 0, stream>>>(obs, obs_emb, obs_aemb, WobsT, obs_b, obs_a, out);

  // Comm conv path + BN stats
  k_conv<<<NSAMP, 256, 0, stream>>>(comm, comm_emb, comm_aemb,
                                    Wc1, Wc3, Wc5, Wc7, cb1, cb3, cb5, cb7,
                                    yg, psumS, psumQ);
  k_bnred<<<H, 256, 0, stream>>>(psumS, psumQ, bnS, bnQ);
  k_bn<<<(NSAMP * LSTEP * H / 8 + 255) / 256, 256, 0, stream>>>(yg, bnS, bnQ, bn_g, bn_b, cnn_a);

  // MTRNN + final linear -> out[:, 256:512]
  k_rnn<<<NSAMP / 16, 256, 0, stream>>>(yg, WxT, WhT, linT, bx, bh, lin_b,
                                        lin_a1, lin_a2, out);
}
